// MoEMixerModel_31026843746797
// MI455X (gfx1250) — matrix operand order repaired and checked by reading, not compiled, not run on hardware
//
#include <hip/hip_runtime.h>

// ---------------------------------------------------------------------------
// Model dims (compile-time constants from the reference)
// ---------------------------------------------------------------------------
#define BATCH    4
#define SEQ      1024
#define DMODEL   512
#define DINNER   1024
#define DSTATE   16
#define DTRANK   32
#define NEXP     4
#define HIDDEN   2048
#define MTOK     (BATCH * SEQ)          // 4096 tokens

typedef __attribute__((ext_vector_type(16))) _Float16 v16h;
typedef __attribute__((ext_vector_type(8)))  _Float16 v8h;
typedef __attribute__((ext_vector_type(8)))  float    v8f;

constexpr int ACT_NONE = 0;
constexpr int ACT_GELU = 2;
constexpr int ACT_SOFTPLUS = 3;

// ---------------------------------------------------------------------------
// Small device helpers
// ---------------------------------------------------------------------------
__device__ __forceinline__ float sigmoidf_(float x) { return 1.0f / (1.0f + __expf(-x)); }

template<int ACT>
__device__ __forceinline__ float apply_act(float x) {
    if constexpr (ACT == ACT_GELU)      return 0.5f * x * (1.0f + erff(x * 0.70710678118654752f));
    else if constexpr (ACT == ACT_SOFTPLUS) return (x > 20.0f) ? x : log1pf(__expf(x));
    else                                return x;
}

__device__ __forceinline__ float wave_sum(float v) {
#pragma unroll
    for (int m = 16; m >= 1; m >>= 1) v += __shfl_xor(v, m, 32);
    return v;
}

// Pack 8 consecutive fp32 into v8h
__device__ __forceinline__ v8h pack8(const float4 a, const float4 b) {
    v8h h;
    h[0]=(_Float16)a.x; h[1]=(_Float16)a.y; h[2]=(_Float16)a.z; h[3]=(_Float16)a.w;
    h[4]=(_Float16)b.x; h[5]=(_Float16)b.y; h[6]=(_Float16)b.z; h[7]=(_Float16)b.w;
    return h;
}

// A fragment (16x32 f16): lane l -> row M = l%16;
// half i -> K = hi*16 + i
__device__ __forceinline__ v16h load_afrag(const _Float16* __restrict__ ar, int hi) {
    union { v16h v; v8h h[2]; } u;
    u.h[0] = *(const v8h*)(ar + hi * 16);
    u.h[1] = *(const v8h*)(ar + hi * 16 + 8);
    return u.v;
}
// B fragment (32x16 f16): lane l -> col N = l%16; half i -> K = hi*16 + i
__device__ __forceinline__ v16h load_bfrag(const _Float16* __restrict__ br) {
    union { v16h v; v8h h[2]; } u;
    u.h[0] = *(const v8h*)(br);
    u.h[1] = *(const v8h*)(br + 8);
    return u.v;
}

// ---------------------------------------------------------------------------
// WMMA GEMM:  C[M,N] = act(A[M,K](row,lda) @ W[N,K]^T + bias[N])
// Optional:   C[m,n] += rowScale[m*rsStride] * val   (MoE gated accumulate)
//
// Block: 256 threads = 8 waves; block tile 128(M) x 64(N); K step 32.
// Wave grid 4(M) x 2(N); each wave owns a 32x32 patch:
//   2 A fragments x 2 B fragments -> 4 v_wmma_f32_16x16x32_f16 per K step,
//   each fragment reused twice (2 ds_load_b128 per WMMA).
// Requires: M%128==0, N%64==0, K%32==0 (true for every GEMM in this model).
// ---------------------------------------------------------------------------
template<int ACT, bool ADD_ROWSCALE>
__global__ __launch_bounds__(256) void gemm_wmma_k(
    const float* __restrict__ A, int lda,
    const float* __restrict__ W,
    const float* __restrict__ bias,
    float* __restrict__ C,
    const float* __restrict__ rowScale, int rsStride,
    int M, int N, int K)
{
    __shared__ _Float16 As[128][40];  // 32 used + pad (rows stay 16B aligned: 80B stride)
    __shared__ _Float16 Ws[64][40];

    const int tid  = threadIdx.x;
    const int lane = tid & 31;
    const int wv   = tid >> 5;
    const int wm   = wv & 3;            // 0..3 -> 32-row slab
    const int wn   = wv >> 2;           // 0..1 -> 32-col slab
    const int m0   = blockIdx.x * 128;
    const int n0   = blockIdx.y * 64;
    const int hi   = lane >> 4;         // half-wave select
    const int l16  = lane & 15;

    // cooperative-load coordinates
    const int arow = tid >> 1;          // 0..127, 16 floats each
    const int acol = (tid & 1) * 16;    // 0 or 16
    const int wrow = tid >> 2;          // 0..63, 8 floats each
    const int wcol = (tid & 3) * 8;     // 0,8,16,24

    v8f acc00 = {}, acc01 = {}, acc10 = {}, acc11 = {};

    for (int k0 = 0; k0 < K; k0 += 32) {
        // ---- cooperative global->LDS (fp32 -> f16)
        {
            const float4* ap = (const float4*)(A + (size_t)(m0 + arow) * lda + k0 + acol);
            float4 a0 = ap[0], a1 = ap[1], a2 = ap[2], a3 = ap[3];
            *(v8h*)&As[arow][acol]     = pack8(a0, a1);
            *(v8h*)&As[arow][acol + 8] = pack8(a2, a3);

            const float4* wp = (const float4*)(W + (size_t)(n0 + wrow) * K + k0 + wcol);
            float4 w0 = wp[0], w1 = wp[1];
            *(v8h*)&Ws[wrow][wcol] = pack8(w0, w1);

            if (k0 + 32 < K) {  // prefetch next K tile into cache
                __builtin_prefetch((const float*)ap + 32, 0, 3);
                __builtin_prefetch((const float*)wp + 32, 0, 3);
            }
        }
        __syncthreads();

        const v16h a0 = load_afrag(&As[32 * wm + l16][0], hi);
        const v16h a1 = load_afrag(&As[32 * wm + 16 + l16][0], hi);
        const v16h b0 = load_bfrag(&Ws[32 * wn + l16][hi * 16]);
        const v16h b1 = load_bfrag(&Ws[32 * wn + 16 + l16][hi * 16]);

        acc00 = __builtin_amdgcn_wmma_f32_16x16x32_f16(false, a0, false, b0, (short)0, acc00, false, false);
        acc01 = __builtin_amdgcn_wmma_f32_16x16x32_f16(false, a0, false, b1, (short)0, acc01, false, false);
        acc10 = __builtin_amdgcn_wmma_f32_16x16x32_f16(false, a1, false, b0, (short)0, acc10, false, false);
        acc11 = __builtin_amdgcn_wmma_f32_16x16x32_f16(false, a1, false, b1, (short)0, acc11, false, false);
        __syncthreads();
    }

    // ---- epilogue: C/D layout -> VGPR j: row = j + hi*8, col = lane%16
    const int r0 = m0 + 32 * wm + hi * 8;       // rows of acc0x
    const int r1 = r0 + 16;                     // rows of acc1x
    const int c0 = n0 + 32 * wn + l16;
    const int c1 = c0 + 16;
    const float b0 = bias ? bias[c0] : 0.0f;
    const float b1 = bias ? bias[c1] : 0.0f;
#pragma unroll
    for (int j = 0; j < 8; j++) {
        const int ra = r0 + j, rb = r1 + j;
        float v00 = apply_act<ACT>(acc00[j] + b0);
        float v01 = apply_act<ACT>(acc01[j] + b1);
        float v10 = apply_act<ACT>(acc10[j] + b0);
        float v11 = apply_act<ACT>(acc11[j] + b1);
        if constexpr (ADD_ROWSCALE) {
            const float ga = rowScale[(size_t)ra * rsStride];
            const float gb = rowScale[(size_t)rb * rsStride];
            C[(size_t)ra * N + c0] += ga * v00;
            C[(size_t)ra * N + c1] += ga * v01;
            C[(size_t)rb * N + c0] += gb * v10;
            C[(size_t)rb * N + c1] += gb * v11;
        } else {
            C[(size_t)ra * N + c0] = v00;
            C[(size_t)ra * N + c1] = v01;
            C[(size_t)rb * N + c0] = v10;
            C[(size_t)rb * N + c1] = v11;
        }
    }
}

// ---------------------------------------------------------------------------
// Elementwise helpers
// ---------------------------------------------------------------------------
__global__ void copy_k(float* __restrict__ o, const float* __restrict__ a, int n) {
    int i = blockIdx.x * blockDim.x + threadIdx.x;
    if (i < n) o[i] = a[i];
}
__global__ void add_k(float* __restrict__ o, const float* __restrict__ a,
                      const float* __restrict__ b, int n) {
    int i = blockIdx.x * blockDim.x + threadIdx.x;
    if (i < n) o[i] = a[i] + b[i];
}
__global__ void add_inplace_k(float* __restrict__ o, const float* __restrict__ a, int n) {
    int i = blockIdx.x * blockDim.x + threadIdx.x;
    if (i < n) o[i] += a[i];
}
__global__ void zero_k(float* __restrict__ p, int n) {
    int i = blockIdx.x * blockDim.x + threadIdx.x;
    if (i < n) p[i] = 0.0f;
}
// reverse along L: out[b,l,:] = in[b,L-1-l,:]
__global__ void reverse_k(const float* __restrict__ in, float* __restrict__ out, int n) {
    int i = blockIdx.x * blockDim.x + threadIdx.x;
    if (i >= n) return;
    int d = i & (DMODEL - 1);
    int l = (i >> 9) & (SEQ - 1);
    int b = i >> 19;
    out[(((size_t)b * SEQ) + l) * DMODEL + d] =
        in[(((size_t)b * SEQ) + (SEQ - 1 - l)) * DMODEL + d];
}

// ---------------------------------------------------------------------------
// LayerNorm over last dim (512): one wave per row, wave32 shuffle reductions
// ---------------------------------------------------------------------------
__global__ __launch_bounds__(256) void layernorm_k(
    const float* __restrict__ x, const float* __restrict__ w,
    const float* __restrict__ b, float* __restrict__ out, int rows)
{
    const int gwave = (blockIdx.x * blockDim.x + threadIdx.x) >> 5;
    const int lane  = threadIdx.x & 31;
    if (gwave >= rows) return;
    const float4* row = (const float4*)(x + (size_t)gwave * DMODEL);
    float4 v[4];
    float s = 0.0f, s2 = 0.0f;
#pragma unroll
    for (int i = 0; i < 4; i++) {
        v[i] = row[lane + 32 * i];
        s  += v[i].x + v[i].y + v[i].z + v[i].w;
        s2 += v[i].x*v[i].x + v[i].y*v[i].y + v[i].z*v[i].z + v[i].w*v[i].w;
    }
    s  = wave_sum(s);
    s2 = wave_sum(s2);
    const float mu  = s * (1.0f / DMODEL);
    const float var = s2 * (1.0f / DMODEL) - mu * mu;
    const float rs  = rsqrtf(var + 1e-5f);
    float4* orow = (float4*)(out + (size_t)gwave * DMODEL);
#pragma unroll
    for (int i = 0; i < 4; i++) {
        const float4 wv = ((const float4*)w)[lane + 32 * i];
        const float4 bv = ((const float4*)b)[lane + 32 * i];
        float4 o;
        o.x = (v[i].x - mu) * rs * wv.x + bv.x;
        o.y = (v[i].y - mu) * rs * wv.y + bv.y;
        o.z = (v[i].z - mu) * rs * wv.z + bv.z;
        o.w = (v[i].w - mu) * rs * wv.w + bv.w;
        orow[lane + 32 * i] = o;
    }
}

// ---------------------------------------------------------------------------
// Depthwise causal conv(width 4) + bias + SiLU.
// xz layout (tok, 2*DINNER): xc = cols [0,DINNER). Output xcs (tok, DINNER).
// ---------------------------------------------------------------------------
__global__ void conv_silu_k(const float* __restrict__ xz, const float* __restrict__ cw,
                            const float* __restrict__ cb, float* __restrict__ out, int n)
{
    int i = blockIdx.x * blockDim.x + threadIdx.x;
    if (i >= n) return;
    const int d  = i & (DINNER - 1);
    const int bl = i >> 10;                 // b*SEQ + l
    const int l  = bl & (SEQ - 1);
    float s = cb[d];
#pragma unroll
    for (int j = 0; j < 4; j++) {
        const int li = l - 3 + j;
        if (li >= 0)
            s += xz[((size_t)(bl - 3 + j)) * (2 * DINNER) + d] * cw[d * 4 + j];
    }
    out[(size_t)bl * DINNER + d] = s * sigmoidf_(s);
}

// ---------------------------------------------------------------------------
// Selective scan, fused with z-gate: y = (scan(x,dt,A,B,C) + x*D) * silu(z)
// grid = (DINNER/256, BATCH); one thread per channel, state in registers,
// per-timestep B/C (32 floats) broadcast through LDS.
// ---------------------------------------------------------------------------
__global__ __launch_bounds__(256) void scan_k(
    const float* __restrict__ xcs, const float* __restrict__ dt,
    const float* __restrict__ dbc, const float* __restrict__ A_log,
    const float* __restrict__ Dp,  const float* __restrict__ xz,
    float* __restrict__ y)
{
    const int b = blockIdx.y;
    const int d = blockIdx.x * 256 + threadIdx.x;
    __shared__ float bc[32];

    float a[DSTATE], h[DSTATE];
#pragma unroll
    for (int n = 0; n < DSTATE; n++) {
        a[n] = -__expf(A_log[(size_t)d * DSTATE + n]);
        h[n] = 0.0f;
    }
    const float Dd = Dp[d];

    for (int t = 0; t < SEQ; t++) {
        __syncthreads();
        if (threadIdx.x < 32)
            bc[threadIdx.x] = dbc[((size_t)(b * SEQ + t)) * 64 + DTRANK + threadIdx.x];
        __syncthreads();
        const size_t tok = (size_t)(b * SEQ + t);
        const float dtv = dt[tok * DINNER + d];
        const float xv  = xcs[tok * DINNER + d];
        float acc = 0.0f;
#pragma unroll
        for (int n = 0; n < DSTATE; n++) {
            const float dA = __expf(dtv * a[n]);
            h[n] = dA * h[n] + (dtv * bc[n]) * xv;
            acc += h[n] * bc[DSTATE + n];
        }
        const float yv = acc + xv * Dd;
        const float zv = xz[tok * (2 * DINNER) + DINNER + d];
        y[tok * DINNER + d] = yv * (zv * sigmoidf_(zv));
    }
}

// ---------------------------------------------------------------------------
// MoE gate: logits(4) -> softmax -> top-2 -> renormalized gates.
// One block (4 waves) per token; wave e computes one expert's 512-dot.
// ---------------------------------------------------------------------------
__global__ __launch_bounds__(128) void moe_gate_k(
    const float* __restrict__ x, const float* __restrict__ gw,
    const float* __restrict__ gb, float* __restrict__ gates)
{
    const int m = blockIdx.x;
    const int e = threadIdx.x >> 5;
    const int lane = threadIdx.x & 31;
    __shared__ float lg[NEXP];
    const float* row = x + (size_t)m * DMODEL;
    const float* wr  = gw + (size_t)e * DMODEL;
    float s = 0.0f;
#pragma unroll
    for (int i = 0; i < 16; i++) {
        const int c = lane + 32 * i;
        s += row[c] * wr[c];
    }
    s = wave_sum(s);
    if (lane == 0) lg[e] = s + gb[e];
    __syncthreads();
    if (threadIdx.x == 0) {
        float mx = lg[0];
        for (int i = 1; i < NEXP; i++) mx = fmaxf(mx, lg[i]);
        float sc[NEXP]; float den = 0.0f;
        for (int i = 0; i < NEXP; i++) { sc[i] = __expf(lg[i] - mx); den += sc[i]; }
        for (int i = 0; i < NEXP; i++) sc[i] /= den;
        int i1 = 0;
        for (int i = 1; i < NEXP; i++) if (sc[i] > sc[i1]) i1 = i;
        int i2 = (i1 == 0) ? 1 : 0;
        for (int i = 0; i < NEXP; i++) if (i != i1 && sc[i] > sc[i2]) i2 = i;
        const float inv = 1.0f / (sc[i1] + sc[i2] + 1e-6f);
        for (int i = 0; i < NEXP; i++)
            gates[(size_t)m * NEXP + i] = (i == i1 || i == i2) ? sc[i] * inv : 0.0f;
    }
}

// ---------------------------------------------------------------------------
// Attention pooling: attn = softmax_L(states @ pw + pb); pool = attn^T @ states
// One block per batch element.
// ---------------------------------------------------------------------------
__global__ __launch_bounds__(256) void pool_k(
    const float* __restrict__ st, const float* __restrict__ pw,
    const float* __restrict__ pb, float* __restrict__ pool)
{
    const int b = blockIdx.x;
    __shared__ float lg[SEQ];
    __shared__ float red[8];
    const size_t rowBase = (size_t)b * SEQ;
    for (int l = threadIdx.x; l < SEQ; l += 256) {
        const float* row = st + (rowBase + l) * DMODEL;
        float s = 0.0f;
        for (int k = 0; k < DMODEL; k++) s += row[k] * pw[k];
        lg[l] = s + pb[0];
    }
    __syncthreads();
    const int lane = threadIdx.x & 31, wv = threadIdx.x >> 5;
    // max
    float mx = -3.0e38f;
    for (int l = threadIdx.x; l < SEQ; l += 256) mx = fmaxf(mx, lg[l]);
#pragma unroll
    for (int m = 16; m >= 1; m >>= 1) mx = fmaxf(mx, __shfl_xor(mx, m, 32));
    if (lane == 0) red[wv] = mx;
    __syncthreads();
    if (threadIdx.x == 0) {
        float v = red[0];
        for (int i = 1; i < 8; i++) v = fmaxf(v, red[i]);
        red[0] = v;
    }
    __syncthreads();
    mx = red[0];
    __syncthreads();
    // exp + sum
    float sum = 0.0f;
    for (int l = threadIdx.x; l < SEQ; l += 256) { const float e = __expf(lg[l] - mx); lg[l] = e; sum += e; }
    sum = wave_sum(sum);
    if (lane == 0) red[wv] = sum;
    __syncthreads();
    if (threadIdx.x == 0) {
        float v = 0.0f;
        for (int i = 0; i < 8; i++) v += red[i];
        red[0] = 1.0f / v;
    }
    __syncthreads();
    const float inv = red[0];
    __syncthreads();
    for (int l = threadIdx.x; l < SEQ; l += 256) lg[l] *= inv;
    __syncthreads();
    // weighted sum over L for each feature
    for (int d = threadIdx.x; d < DMODEL; d += 256) {
        float acc = 0.0f;
        for (int l = 0; l < SEQ; l++) acc += lg[l] * st[(rowBase + l) * DMODEL + d];
        pool[(size_t)b * DMODEL + d] = acc;
    }
}

// out[b,j] = [f_pool | b_pool] @ LL_w^T + LL_b   (LL_w: 512 x 1024)
__global__ void final_linear_k(const float* __restrict__ fp, const float* __restrict__ bp,
                               const float* __restrict__ W, const float* __restrict__ bias,
                               float* __restrict__ out)
{
    int i = blockIdx.x * blockDim.x + threadIdx.x;
    if (i >= BATCH * DMODEL) return;
    const int b = i >> 9, j = i & (DMODEL - 1);
    const float* wr = W + (size_t)j * (2 * DMODEL);
    float s = bias[j];
    for (int k = 0; k < DMODEL; k++) s += fp[(size_t)b * DMODEL + k] * wr[k];
    for (int k = 0; k < DMODEL; k++) s += bp[(size_t)b * DMODEL + k] * wr[DMODEL + k];
    out[i] = s;
}

// ---------------------------------------------------------------------------
// Host side
// ---------------------------------------------------------------------------
struct LayerP {
    const float *norm_w, *norm_b, *in_proj_w, *in_proj_b, *conv_w, *conv_b, *x_proj_w,
                *dt_proj_w, *dt_proj_b, *A_log, *D, *out_proj_w, *gate_w, *gate_b,
                *w1, *b1, *w2, *b2;
};
static LayerP load_layer(void* const* d, int base) {
    LayerP p;
    p.norm_w     = (const float*)d[base + 0];
    p.norm_b     = (const float*)d[base + 1];
    p.in_proj_w  = (const float*)d[base + 2];
    p.in_proj_b  = (const float*)d[base + 3];
    p.conv_w     = (const float*)d[base + 4];
    p.conv_b     = (const float*)d[base + 5];
    p.x_proj_w   = (const float*)d[base + 6];
    p.dt_proj_w  = (const float*)d[base + 7];
    p.dt_proj_b  = (const float*)d[base + 8];
    p.A_log      = (const float*)d[base + 9];
    p.D          = (const float*)d[base + 10];
    p.out_proj_w = (const float*)d[base + 11];
    p.gate_w     = (const float*)d[base + 12];
    p.gate_b     = (const float*)d[base + 13];
    p.w1         = (const float*)d[base + 14];
    p.b1         = (const float*)d[base + 15];
    p.w2         = (const float*)d[base + 16];
    p.b2         = (const float*)d[base + 17];
    return p;
}

static inline dim3 egrid(int n) { return dim3((n + 255) / 256); }

static void launch_gemm(hipStream_t s, const float* A, int lda, const float* W,
                        const float* bias, float* C, int M, int N, int K,
                        int act, const float* rowScale, int rsStride)
{
    dim3 g(M / 128, N / 64), b(256);
    if (rowScale) {
        gemm_wmma_k<ACT_NONE, true><<<g, b, 0, s>>>(A, lda, W, bias, C, rowScale, rsStride, M, N, K);
    } else if (act == ACT_GELU) {
        gemm_wmma_k<ACT_GELU, false><<<g, b, 0, s>>>(A, lda, W, bias, C, nullptr, 0, M, N, K);
    } else if (act == ACT_SOFTPLUS) {
        gemm_wmma_k<ACT_SOFTPLUS, false><<<g, b, 0, s>>>(A, lda, W, bias, C, nullptr, 0, M, N, K);
    } else {
        gemm_wmma_k<ACT_NONE, false><<<g, b, 0, s>>>(A, lda, W, bias, C, nullptr, 0, M, N, K);
    }
}

struct Scratch {
    float *resid, *states, *normed, *curr, *xz, *xcs, *dbc, *dtb, *ybuf, *gates, *hbuf, *moe;
};

static void run_direction(hipStream_t s, const float* hs, const LayerP* Ls, Scratch& w, float* out)
{
    const int NE = MTOK * DMODEL;
    copy_k<<<egrid(NE), 256, 0, s>>>(w.resid, hs, NE);            // residual = hs (first states)
    for (int li = 0; li < 2; li++) {
        const LayerP& p = Ls[li];
        if (li > 0) add_inplace_k<<<egrid(NE), 256, 0, s>>>(w.resid, w.states, NE);
        layernorm_k<<<MTOK / 8, 256, 0, s>>>(w.resid, p.norm_w, p.norm_b, w.normed, MTOK);
        // ---- mamba mixer
        launch_gemm(s, w.normed, DMODEL, p.in_proj_w, p.in_proj_b, w.xz,
                    MTOK, 2 * DINNER, DMODEL, ACT_NONE, nullptr, 0);
        conv_silu_k<<<egrid(MTOK * DINNER), 256, 0, s>>>(w.xz, p.conv_w, p.conv_b, w.xcs,
                                                         MTOK * DINNER);
        launch_gemm(s, w.xcs, DINNER, p.x_proj_w, nullptr, w.dbc,
                    MTOK, DTRANK + 2 * DSTATE, DINNER, ACT_NONE, nullptr, 0);
        launch_gemm(s, w.dbc, DTRANK + 2 * DSTATE, p.dt_proj_w, p.dt_proj_b, w.dtb,
                    MTOK, DINNER, DTRANK, ACT_SOFTPLUS, nullptr, 0);
        scan_k<<<dim3(DINNER / 256, BATCH), 256, 0, s>>>(w.xcs, w.dtb, w.dbc,
                                                         p.A_log, p.D, w.xz, w.ybuf);
        launch_gemm(s, w.ybuf, DINNER, p.out_proj_w, nullptr, w.states,
                    MTOK, DMODEL, DINNER, ACT_NONE, nullptr, 0);
        // ---- residual & MoE
        add_k<<<egrid(NE), 256, 0, s>>>(w.curr, w.states, w.resid, NE);
        moe_gate_k<<<MTOK, 128, 0, s>>>(w.curr, p.gate_w, p.gate_b, w.gates);
        zero_k<<<egrid(NE), 256, 0, s>>>(w.moe, NE);
        for (int e = 0; e < NEXP; e++) {
            launch_gemm(s, w.curr, DMODEL, p.w1 + (size_t)e * HIDDEN * DMODEL,
                        p.b1 + (size_t)e * HIDDEN, w.hbuf,
                        MTOK, HIDDEN, DMODEL, ACT_GELU, nullptr, 0);
            launch_gemm(s, w.hbuf, HIDDEN, p.w2 + (size_t)e * DMODEL * HIDDEN,
                        p.b2 + (size_t)e * DMODEL, w.moe,
                        MTOK, DMODEL, HIDDEN, ACT_NONE, w.gates + e, NEXP);
        }
        add_inplace_k<<<egrid(NE), 256, 0, s>>>(w.resid, w.moe, NE);
    }
    add_k<<<egrid(NE), 256, 0, s>>>(out, w.states, w.resid, NE);
}

extern "C" void kernel_launch(void* const* d_in, const int* in_sizes, int n_in,
                              void* d_out, int out_size, void* d_ws, size_t ws_size,
                              hipStream_t stream)
{
    (void)in_sizes; (void)n_in; (void)out_size; (void)ws_size;
    const float* x = (const float*)d_in[0];
    LayerP fwd[2], bwd[2];
    for (int i = 0; i < 2; i++) fwd[i] = load_layer(d_in, 1 + 18 * i);
    for (int i = 0; i < 2; i++) bwd[i] = load_layer(d_in, 1 + 36 + 18 * i);
    const float* norm_f_w = (const float*)d_in[73];
    const float* norm_f_b = (const float*)d_in[74];
    const float* fpool_w  = (const float*)d_in[75];
    const float* fpool_b  = (const float*)d_in[76];
    const float* bpool_w  = (const float*)d_in[77];
    const float* bpool_b  = (const float*)d_in[78];
    const float* LL_w     = (const float*)d_in[79];
    const float* LL_b     = (const float*)d_in[80];

    // ---- workspace carve-up
    float* ws = (float*)d_ws;
    size_t off = 0;
    auto alloc = [&](size_t n) { float* p = ws + off; off += n; return p; };
    Scratch S;
    float* revx    = alloc((size_t)MTOK * DMODEL);
    S.resid        = alloc((size_t)MTOK * DMODEL);
    S.states       = alloc((size_t)MTOK * DMODEL);
    S.normed       = alloc((size_t)MTOK * DMODEL);
    S.curr         = alloc((size_t)MTOK * DMODEL);
    float* fres    = alloc((size_t)MTOK * DMODEL);
    float* bres    = alloc((size_t)MTOK * DMODEL);
    float* fstates = alloc((size_t)MTOK * DMODEL);
    float* bstates = alloc((size_t)MTOK * DMODEL);
    S.xz           = alloc((size_t)MTOK * 2 * DINNER);
    S.xcs          = alloc((size_t)MTOK * DINNER);
    S.dbc          = alloc((size_t)MTOK * 64);
    S.dtb          = alloc((size_t)MTOK * DINNER);
    S.ybuf         = alloc((size_t)MTOK * DINNER);
    S.gates        = alloc((size_t)MTOK * NEXP);
    S.moe          = alloc((size_t)MTOK * DMODEL);
    float* fpool   = alloc((size_t)BATCH * DMODEL);
    float* bpool   = alloc((size_t)BATCH * DMODEL);
    S.hbuf         = S.xz;   // alias: xz is dead once the scan consumed z

    // ---- forward direction
    run_direction(stream, x, fwd, S, fres);
    // ---- backward direction on reversed sequence
    reverse_k<<<egrid(MTOK * DMODEL), 256, 0, stream>>>(x, revx, MTOK * DMODEL);
    run_direction(stream, revx, bwd, S, bres);

    // ---- final norms, pooling, head
    layernorm_k<<<MTOK / 8, 256, 0, stream>>>(fres, norm_f_w, norm_f_b, fstates, MTOK);
    layernorm_k<<<MTOK / 8, 256, 0, stream>>>(bres, norm_f_w, norm_f_b, bstates, MTOK);
    pool_k<<<BATCH, 256, 0, stream>>>(fstates, fpool_w, fpool_b, fpool);
    pool_k<<<BATCH, 256, 0, stream>>>(bstates, bpool_w, bpool_b, bpool);
    final_linear_k<<<egrid(BATCH * DMODEL), 256, 0, stream>>>(fpool, bpool, LL_w, LL_b,
                                                              (float*)d_out);
}
